// LayerNormGRU_22677427322908
// MI455X (gfx1250) — matrix operand order repaired and checked by reading, not compiled, not run on hardware
//
#include <hip/hip_runtime.h>
#include <hip/hip_bf16.h>
#include <math.h>

// Problem constants
static constexpr int kH  = 512;
static constexpr int kT  = 128;
static constexpr int kB  = 512;
static constexpr int kIn = 512;
static constexpr int kG  = 3 * kH;                    // 1536 gate columns
static constexpr long long kTB = (long long)kT * kB;  // 65536 rows

typedef __bf16 bf16;
typedef __attribute__((ext_vector_type(16))) __bf16 v16bf;
typedef __attribute__((ext_vector_type(8)))  __bf16 v8bf;
typedef __attribute__((ext_vector_type(4)))  __bf16 v4bf;
typedef __attribute__((ext_vector_type(8)))  float   v8f;
typedef __attribute__((ext_vector_type(4)))  float   v4f;

// ---------------------------------------------------------------------------
// A-fragment (16x32 bf16, M x K) from a bf16 source: lane l holds row m = l&15,
// K chunk [k0 + 8*(l>>4), +8) in VGPRs 0-3 and [k0 + 16 + 8*(l>>4), +8) in 4-7.
static __device__ __forceinline__ v16bf load_a_frag(const bf16* __restrict__ base,
                                                    int ld, int k0, int lane) {
    const int m  = lane & 15;
    const int hk = (lane >> 4) & 1;
    const bf16* p = base + (size_t)m * ld + k0 + 8 * hk;
    v8bf lo = *(const v8bf*)p;
    v8bf hi = *(const v8bf*)(p + 16);
    return __builtin_shufflevector(lo, hi, 0,1,2,3,4,5,6,7,8,9,10,11,12,13,14,15);
}

// Same fragment built directly from an fp32 source (fused RNE convert).
// xs is consumed exactly once, so converting in-register beats a staging pass.
static __device__ __forceinline__ v16bf load_a_frag_f32(const float* __restrict__ base,
                                                        int ld, int k0, int lane) {
    const int m  = lane & 15;
    const int hk = (lane >> 4) & 1;
    const float* p = base + (size_t)m * ld + k0 + 8 * hk;
    v4f f0 = *(const v4f*)(p);
    v4f f1 = *(const v4f*)(p + 4);
    v4f f2 = *(const v4f*)(p + 16);
    v4f f3 = *(const v4f*)(p + 20);
    v16bf a;
    #pragma unroll
    for (int i = 0; i < 4; ++i) {
        a[i]      = (bf16)f0[i];
        a[4 + i]  = (bf16)f1[i];
        a[8 + i]  = (bf16)f2[i];
        a[12 + i] = (bf16)f3[i];
    }
    return a;
}

// B-fragment (32x16 bf16, K x N) with B = W^T, W row-major [n][k]:
// lane l holds column n = l&15, K chunk [k0 + 8*(l>>4), +8) in 0-7 and [k0 + 16 + 8*(l>>4), +8) in 8-15.
static __device__ __forceinline__ v16bf load_b_frag(const bf16* __restrict__ w,
                                                    int ld, int n0, int k0, int lane) {
    const int n  = lane & 15;
    const int hk = (lane >> 4) & 1;
    const bf16* p = w + (size_t)(n0 + n) * ld + k0 + 8 * hk;
    v8bf lo = *(const v8bf*)p;
    v8bf hi = *(const v8bf*)(p + 16);
    return __builtin_shufflevector(lo, hi, 0,1,2,3,4,5,6,7,8,9,10,11,12,13,14,15);
}

// ---------------------------------------------------------------------------
// CDNA5 async copy: global -> LDS, 16 bytes per lane, tracked by ASYNCcnt.
// lds_off is the DS-relative address (low 32 bits of the generic LDS pointer).
static __device__ __forceinline__ void async_copy_b128(unsigned lds_off, const void* gptr) {
    asm volatile("global_load_async_to_lds_b128 %0, %1, off"
                 :: "v"(lds_off), "v"((unsigned long long)(uintptr_t)gptr)
                 : "memory");
}
static __device__ __forceinline__ void wait_async_zero() {
    asm volatile("s_wait_asynccnt 0x0" ::: "memory");
}

// ---------------------------------------------------------------------------
// fp32 -> bf16 conversion (RNE), 4 elements/thread (weights only)
__global__ void cvt_f32_bf16(const float* __restrict__ src, bf16* __restrict__ dst, int n) {
    int i = (blockIdx.x * blockDim.x + threadIdx.x) * 4;
    if (i + 3 < n) {
        dst[i + 0] = (bf16)src[i + 0];
        dst[i + 1] = (bf16)src[i + 1];
        dst[i + 2] = (bf16)src[i + 2];
        dst[i + 3] = (bf16)src[i + 3];
    }
}

// ---------------------------------------------------------------------------
// Fused GEMM + LayerNorm for the input projections:
//   ih[row, 0:1536] = LN_groups( x[row,:] @ W_ih^T ) * gamma + beta
// Group 0 = cols [0,1024) (rz), group 1 = cols [1024,1536) (n); var with ddof=1.
// 32 rows per WG; 16 waves, each owning 2 row-tiles x 6 col-tiles; B fragments
// reused across both row-tiles; A converted fp32->bf16 in-register.
__global__ __launch_bounds__(512)
void gemm_ih_ln(const float* __restrict__ Xs,      // [65536, 512] fp32
                const bf16* __restrict__ Wbf,      // [1536, 512]  bf16
                const float* __restrict__ gamma,   // [1536]
                const float* __restrict__ beta,    // [1536]
                bf16* __restrict__ ih_out)         // [65536, 1536] bf16
{
    const int row0 = blockIdx.x * 32;
    const int tid  = threadIdx.x;
    const int lane = tid & 31;
    const int wave = tid >> 5;                 // 0..15
    const int hk   = (lane >> 4) & 1;

    __shared__ float s_sum[2][32];
    __shared__ float s_sqs[2][32];
    __shared__ float s_mean[2][32];
    __shared__ float s_rstd[2][32];

    if (tid < 64) { s_sum[tid >> 5][tid & 31] = 0.f; s_sqs[tid >> 5][tid & 31] = 0.f; }
    __syncthreads();

    v8f acc[2][6] = {};
    const float* arow = Xs + (size_t)row0 * kIn;

    for (int k0 = 0; k0 < kIn; k0 += 32) {
        v16bf a0 = load_a_frag_f32(arow,            kIn, k0, lane);
        v16bf a1 = load_a_frag_f32(arow + 16 * kIn, kIn, k0, lane);
        #pragma unroll
        for (int i = 0; i < 6; ++i) {
            const int n0 = (wave * 6 + i) * 16;
            v16bf b = load_b_frag(Wbf, kIn, n0, k0, lane);
            acc[0][i] = __builtin_amdgcn_wmma_f32_16x16x32_bf16(
                false, a0, false, b, (short)0, acc[0][i], false, false);
            acc[1][i] = __builtin_amdgcn_wmma_f32_16x16x32_bf16(
                false, a1, false, b, (short)0, acc[1][i], false, false);
        }
    }

    // Row sums / sums-of-squares. C layout: VGPR v -> row v + 8*hk, col n0 + (lane&15).
    #pragma unroll
    for (int rt = 0; rt < 2; ++rt) {
        #pragma unroll
        for (int i = 0; i < 6; ++i) {
            const int n0  = (wave * 6 + i) * 16;
            const int grp = (n0 < 2 * kH) ? 0 : 1;
            #pragma unroll
            for (int v = 0; v < 8; ++v) {
                const int rr = rt * 16 + v + 8 * hk;
                const float val = acc[rt][i][v];
                atomicAdd(&s_sum[grp][rr], val);
                atomicAdd(&s_sqs[grp][rr], val * val);
            }
        }
    }
    __syncthreads();

    if (tid < 64) {
        const int g = tid >> 5, r = tid & 31;
        const float cnt  = g ? (float)kH : (float)(2 * kH);
        const float mean = s_sum[g][r] / cnt;
        const float var  = (s_sqs[g][r] - cnt * mean * mean) / (cnt - 1.f);  // ddof=1
        s_mean[g][r] = mean;
        s_rstd[g][r] = 1.f / sqrtf(var);    // EPS == 0
    }
    __syncthreads();

    #pragma unroll
    for (int rt = 0; rt < 2; ++rt) {
        #pragma unroll
        for (int i = 0; i < 6; ++i) {
            const int n0  = (wave * 6 + i) * 16;
            const int grp = (n0 < 2 * kH) ? 0 : 1;
            const int col = n0 + (lane & 15);
            const float gc = gamma[col], bc = beta[col];
            #pragma unroll
            for (int v = 0; v < 8; ++v) {
                const int rr = rt * 16 + v + 8 * hk;
                const float val = (acc[rt][i][v] - s_mean[grp][rr]) * s_rstd[grp][rr] * gc + bc;
                ih_out[(size_t)(row0 + rr) * kG + col] = (bf16)val;
            }
        }
    }
}

// ---------------------------------------------------------------------------
// Persistent GRU scan. The recurrence is independent per batch row: each WG
// owns 16 batch rows and iterates all 128 timesteps with h resident in LDS.
// Each step's 48KB ih tile is async-copied into LDS at step start so the DMA
// overlaps the WMMA GEMM; s_wait_asynccnt + barrier gates the gate phase.
__global__ __launch_bounds__(512)
void gru_scan(const float* __restrict__ h0,        // [512, 512]
              const bf16*  __restrict__ Whh,       // [1536, 512] bf16
              const float* __restrict__ gamma_hh,  // [1536]
              const float* __restrict__ beta_hh,   // [1536]
              const bf16*  __restrict__ ih_act,    // [65536, 1536] bf16
              float* __restrict__ out)             // ys [128,512,512] ++ h_last [512,512]
{
    const int r0   = blockIdx.x * 16;              // batch rows owned by this WG
    const int tid  = threadIdx.x;
    const int lane = tid & 31;
    const int wave = tid >> 5;                     // 0..15
    const int hk   = (lane >> 4) & 1;

    __shared__ __align__(32) bf16 h_bf[16][kH];    // 16 KB: bf16 h (WMMA A operand)
    __shared__ __align__(16) float h_f[16][kH];    // 32 KB: fp32 h (gating term)
    __shared__ __align__(16) bf16 hh_buf[16][kG];  // 48 KB: normalized hh pre-activations
    __shared__ __align__(16) bf16 ih_buf[16 * kG]; // 48 KB: async-staged ih tile
    __shared__ float s_sum[2][16];
    __shared__ float s_mean[2][16];

    const unsigned ih_lds_base = (unsigned)(uintptr_t)&ih_buf[0];

    // Load initial hidden state
    for (int i = tid; i < 16 * kH; i += blockDim.x) {
        const int r = i >> 9, k = i & (kH - 1);
        const float v = h0[(size_t)(r0 + r) * kH + k];
        h_f[r][k]  = v;
        h_bf[r][k] = (bf16)v;
    }
    __syncthreads();

    for (int t = 0; t < kT; ++t) {
        // Kick off the async DMA of this step's ih tile (16 x 1536 bf16 = 48 KB);
        // it runs concurrently with the whole GEMM phase below.
        {
            const bf16* src = ih_act + ((size_t)t * kB + r0) * kG;
            for (int i = tid; i < (16 * kG) / 8; i += blockDim.x)   // 16B per lane-op
                async_copy_b128(ih_lds_base + i * 16, src + i * 8);
        }

        if (tid < 32) s_sum[tid >> 4][tid & 15] = 0.f;
        __syncthreads();

        // hh = h @ W_hh^T  (A from LDS, B from L2-resident bf16 weights)
        v8f acc[6] = {};
        for (int k0 = 0; k0 < kH; k0 += 32) {
            v16bf a = load_a_frag(&h_bf[0][0], kH, k0, lane);
            #pragma unroll
            for (int i = 0; i < 6; ++i) {
                const int n0 = (wave * 6 + i) * 16;
                v16bf b = load_b_frag(Whh, kH, n0, k0, lane);
                acc[i] = __builtin_amdgcn_wmma_f32_16x16x32_bf16(
                    false, a, false, b, (short)0, acc[i], false, false);
            }
        }

        // Row means (LN_h subtracts mean only — no variance)
        #pragma unroll
        for (int i = 0; i < 6; ++i) {
            const int n0  = (wave * 6 + i) * 16;
            const int grp = (n0 < 2 * kH) ? 0 : 1;
            #pragma unroll
            for (int v = 0; v < 8; ++v)
                atomicAdd(&s_sum[grp][v + 8 * hk], acc[i][v]);
        }
        __syncthreads();
        if (tid < 32) {
            const int g = tid >> 4, r = tid & 15;
            s_mean[g][r] = s_sum[g][r] / (g ? (float)kH : (float)(2 * kH));
        }
        __syncthreads();

        // Normalize + scale/shift into LDS so gates can pair r/z/n columns
        #pragma unroll
        for (int i = 0; i < 6; ++i) {
            const int n0  = (wave * 6 + i) * 16;
            const int grp = (n0 < 2 * kH) ? 0 : 1;
            const int col = n0 + (lane & 15);
            const float gc = gamma_hh[col], bc = beta_hh[col];
            #pragma unroll
            for (int v = 0; v < 8; ++v) {
                const int rr = v + 8 * hk;
                hh_buf[rr][col] = (bf16)((acc[i][v] - s_mean[grp][rr]) * gc + bc);
            }
        }

        wait_async_zero();      // this wave's ih DMA complete
        __syncthreads();        // all waves' DMA + hh_buf visible

        // Gates + state update; vector LDS reads, 16B stores of ys[t]
        for (int c = tid; c < (16 * kH) / 4; c += blockDim.x) {
            const int r  = c >> 7;              // 128 chunks of 4 per row
            const int j4 = (c & 127) * 4;
            const bf16* ihr = &ih_buf[r * kG];
            v4bf ir4 = *(const v4bf*)(ihr + j4);
            v4bf iz4 = *(const v4bf*)(ihr + kH + j4);
            v4bf in4 = *(const v4bf*)(ihr + 2 * kH + j4);
            v4bf hr4 = *(const v4bf*)(&hh_buf[r][j4]);
            v4bf hz4 = *(const v4bf*)(&hh_buf[r][kH + j4]);
            v4bf hn4 = *(const v4bf*)(&hh_buf[r][2 * kH + j4]);
            v4f  hp4 = *(const v4f*)(&h_f[r][j4]);
            v4f  o4;
            v4bf ob4;
            #pragma unroll
            for (int u = 0; u < 4; ++u) {
                const float rg = 1.f / (1.f + __expf(-((float)ir4[u] + (float)hr4[u])));
                const float zg = 1.f / (1.f + __expf(-((float)iz4[u] + (float)hz4[u])));
                const float ng = tanhf((float)in4[u] + rg * (float)hn4[u]);
                const float hn = (1.f - zg) * ng + zg * hp4[u];
                o4[u]  = hn;
                ob4[u] = (bf16)hn;
            }
            *(v4f*)(&h_f[r][j4])  = o4;
            *(v4bf*)(&h_bf[r][j4]) = ob4;
            *(v4f*)(out + ((size_t)t * kB + (r0 + r)) * kH + j4) = o4;
            if (t == kT - 1)
                *(v4f*)(out + (size_t)kT * kB * kH + (size_t)(r0 + r) * kH + j4) = o4;
        }
        __syncthreads();
    }
}

// ---------------------------------------------------------------------------
extern "C" void kernel_launch(void* const* d_in, const int* in_sizes, int n_in,
                              void* d_out, int out_size, void* d_ws, size_t ws_size,
                              hipStream_t stream) {
    const float* xs   = (const float*)d_in[0];   // [128,512,512]
    const float* h0   = (const float*)d_in[1];   // [1,512,512]
    const float* w_ih = (const float*)d_in[2];   // [1536,512]
    const float* w_hh = (const float*)d_in[3];   // [1536,512]
    const float* b_ih = (const float*)d_in[4];   // [1536]
    const float* b_hh = (const float*)d_in[5];   // [1536]
    const float* g_ih = (const float*)d_in[6];   // [1536]
    const float* g_hh = (const float*)d_in[7];   // [1536]
    float* out = (float*)d_out;

    // Workspace layout (bf16): W_ih | W_hh | ih activations  (~195 MB)
    bf16* Wih_bf = (bf16*)d_ws;
    bf16* Whh_bf = Wih_bf + (size_t)kG * kIn;
    bf16* IH     = Whh_bf + (size_t)kG * kH;

    const int nW = kG * kIn;                 // 786432
    cvt_f32_bf16<<<nW / 1024, 256, 0, stream>>>(w_ih, Wih_bf, nW);
    cvt_f32_bf16<<<nW / 1024, 256, 0, stream>>>(w_hh, Whh_bf, nW);

    gemm_ih_ln<<<(int)(kTB / 32), 512, 0, stream>>>(xs, Wih_bf, g_ih, b_ih, IH);

    gru_scan<<<kB / 16, 512, 0, stream>>>(h0, Whh_bf, g_hh, b_hh, IH, out);
}
